// GIN_GRU_73023033967288
// MI455X (gfx1250) — compile-verified, matrix operand order checked by reading, not run on hardware
//
#include <hip/hip_runtime.h>
#include <hip/hip_bf16.h>
#include <stdint.h>

#define NND 50000
#define NED 800000
#define NGR 256

typedef __attribute__((ext_vector_type(16))) __bf16 v16bf;
typedef __attribute__((ext_vector_type(8)))  float  v8f;

struct Frag { uint4 lo, hi; };

__device__ __forceinline__ unsigned short f2bf(float f) {
    unsigned int u = __float_as_uint(f);
    u += 0x7FFFu + ((u >> 16) & 1u);
    return (unsigned short)(u >> 16);
}

__device__ __forceinline__ unsigned int fenc(float f) {
    unsigned int u = __float_as_uint(f);
    return (u & 0x80000000u) ? ~u : (u | 0x80000000u);
}
__device__ __forceinline__ float fdec(unsigned int u) {
    unsigned int v = (u & 0x80000000u) ? (u & 0x7FFFFFFFu) : ~u;
    return __uint_as_float(v);
}

// ---------------- elementwise helpers ----------------

__global__ void k_copy_f32(const float* __restrict__ a, float* __restrict__ b, int n) {
    int i = blockIdx.x * blockDim.x + threadIdx.x;
    if (i < n) b[i] = a[i];
}

__global__ void k_f2bf(const float* __restrict__ a, unsigned short* __restrict__ b, int n) {
    int i = blockIdx.x * blockDim.x + threadIdx.x;
    if (i < n) b[i] = f2bf(a[i]);
}

// W[in,out] (row-major) -> Wt[out,in] bf16
__global__ void k_transconv(const float* __restrict__ W, unsigned short* __restrict__ Wt,
                            int rows_in, int cols_out) {
    int i = blockIdx.x * blockDim.x + threadIdx.x;
    if (i >= rows_in * cols_out) return;
    int o = i % cols_out, in_ = i / cols_out;
    Wt[o * rows_in + in_] = f2bf(W[i]);
}

__global__ void k_grubias(const float* __restrict__ bih, const float* __restrict__ bhh,
                          float* __restrict__ cb) {
    int j = blockIdx.x * blockDim.x + threadIdx.x;
    if (j < 384) cb[j] = bih[j] + ((j < 256) ? bhh[j] : 0.0f);
}

__global__ void k_zero_f32(float* __restrict__ p, int n) {
    int i = blockIdx.x * blockDim.x + threadIdx.x;
    if (i < n) p[i] = 0.0f;
}

__global__ void k_fill_u32(unsigned int* __restrict__ p, unsigned int v, int n) {
    int i = blockIdx.x * blockDim.x + threadIdx.x;
    if (i < n) p[i] = v;
}

// ---------------- graph ops ----------------

// agg[dst[e]*128+f] += z[src[e]*128+f]   (feature width always 128 for conv inputs)
__global__ void k_scatter(const float* __restrict__ z, const int* __restrict__ src,
                          const int* __restrict__ dst, float* __restrict__ agg) {
    long long i = (long long)blockIdx.x * blockDim.x + threadIdx.x;
    if (i >= (long long)NED * 128) return;
    int e = (int)(i >> 7), f = (int)(i & 127);
    atomicAdd(&agg[(long long)dst[e] * 128 + f], z[(long long)src[e] * 128 + f]);
}

__global__ void k_segmax(const float* __restrict__ z, const int* __restrict__ batch,
                         unsigned int* __restrict__ gmax, int M) {
    int i = blockIdx.x * blockDim.x + threadIdx.x;
    if (i >= M * 32) return;
    int node = i >> 5, c = i & 31;
    atomicMax(&gmax[batch[node] * 32 + c], fenc(z[node * 32 + c]));
}

__global__ void k_gdecode(const unsigned int* __restrict__ gmax, float* __restrict__ g,
                          unsigned short* __restrict__ gbf) {
    int i = blockIdx.x * blockDim.x + threadIdx.x;
    if (i >= NGR * 32) return;
    float v = fdec(gmax[i]);
    if (!isfinite(v)) v = 0.0f;
    g[i] = v; gbf[i] = f2bf(v);
}

// ---------------- BatchNorm ----------------

// stats[0:C) = sum, stats[128:128+C) = sumsq  (C power of 2, <=128)
__global__ void k_bnreduce(const float* __restrict__ H, float* __restrict__ stats,
                           int M, int C) {
    __shared__ float ssum[128], ssq[128];
    int t = threadIdx.x;
    if (t < C) { ssum[t] = 0.0f; ssq[t] = 0.0f; }
    __syncthreads();
    int cm = C - 1;
    long long n = (long long)M * C;
    for (long long i = (long long)blockIdx.x * blockDim.x + t; i < n;
         i += (long long)gridDim.x * blockDim.x) {
        float v = H[i];
        int c = (int)(i & cm);
        atomicAdd(&ssum[c], v);
        atomicAdd(&ssq[c], v * v);
    }
    __syncthreads();
    if (t < C) { atomicAdd(&stats[t], ssum[t]); atomicAdd(&stats[128 + t], ssq[t]); }
}

__global__ void k_bnscale(const float* __restrict__ stats, const float* __restrict__ gamma,
                          const float* __restrict__ beta, float* __restrict__ scsh,
                          int M, int C) {
    int c = threadIdx.x;
    if (c >= C) return;
    float mean = stats[c] / (float)M;
    float var  = stats[128 + c] / (float)M - mean * mean;
    float sc = gamma[c] * rsqrtf(var + 1e-5f);
    scsh[c] = sc;
    scsh[128 + c] = beta[c] - mean * sc;
}

__global__ void k_bnapply(const float* __restrict__ H, const float* __restrict__ scsh,
                          float* __restrict__ z, long long n, int cm) {
    long long i = (long long)blockIdx.x * blockDim.x + threadIdx.x;
    if (i >= n) return;
    int c = (int)(i & cm);
    z[i] = H[i] * scsh[c] + scsh[128 + c];
}

// ---------------- WMMA GEMM ----------------
// C[M,N] = act(A[M,K](bf16) * Bt[N,K](bf16)^T + bias[N])
// M,N multiples of 16; K multiple of 32. 8 waves/block, one 16x16 tile each.
template <int RELU, int OUTBF>
__global__ void k_gemm(const unsigned short* __restrict__ A,
                       const unsigned short* __restrict__ Bt,
                       const float* __restrict__ bias,
                       float* __restrict__ Cf, unsigned short* __restrict__ Cb,
                       int M, int N, int K) {
    int wave = threadIdx.x >> 5;
    int lane = threadIdx.x & 31;
    int ntiles = N >> 4, mtiles = M >> 4;
    int tile = blockIdx.x * 8 + wave;
    if (tile >= mtiles * ntiles) return;
    int mt = tile / ntiles, nt = tile % ntiles;
    int r  = lane & 15;        // row of A / row of Bt (= col of B)
    int kg = lane >> 4;        // K-group: 0 -> K{0..7,16..23}, 1 -> K{8..15,24..31}

    const unsigned short* Arow = A  + (long long)(mt * 16 + r) * K + kg * 8;
    const unsigned short* Brow = Bt + (long long)(nt * 16 + r) * K + kg * 8;

    v8f acc = {};
    for (int kk = 0; kk < K; kk += 32) {
        Frag fa, fb;
        fa.lo = *(const uint4*)(Arow + kk);
        fa.hi = *(const uint4*)(Arow + kk + 16);
        fb.lo = *(const uint4*)(Brow + kk);
        fb.hi = *(const uint4*)(Brow + kk + 16);
        v16bf a = __builtin_bit_cast(v16bf, fa);
        v16bf b = __builtin_bit_cast(v16bf, fb);
        acc = __builtin_amdgcn_wmma_f32_16x16x32_bf16(
            false, a, false, b, (short)0, acc, false, false);
    }

    int col = nt * 16 + r;
    float bv = bias ? bias[col] : 0.0f;
    int row0 = mt * 16 + kg * 8;
#pragma unroll
    for (int v = 0; v < 8; v++) {
        float val = acc[v] + bv;
        if (RELU) val = fmaxf(val, 0.0f);
        long long idx = (long long)(row0 + v) * N + col;
        if (OUTBF) Cb[idx] = f2bf(val); else Cf[idx] = val;
    }
}

// ---------------- GRU gates / final ----------------

__global__ void k_gru_gate(const float* __restrict__ gi, const float* __restrict__ bhh,
                           float* __restrict__ h, unsigned short* __restrict__ hbf) {
    int i = blockIdx.x * blockDim.x + threadIdx.x;
    if (i >= NGR * 128) return;
    int g = i >> 7, j = i & 127;
    const float* row = gi + g * 384;
    float r  = 1.0f / (1.0f + expf(-row[j]));
    float zg = 1.0f / (1.0f + expf(-row[128 + j]));
    float n  = tanhf(row[256 + j] + r * bhh[256 + j]);
    float hv = (1.0f - zg) * n;
    h[i] = hv; hbf[i] = f2bf(hv);
}

__global__ void k_final(const float* __restrict__ y3, const float* __restrict__ wl,
                        const float* __restrict__ bl, float* __restrict__ out) {
    int g = blockIdx.x * blockDim.x + threadIdx.x;
    if (g >= NGR) return;
    float s = bl[0];
    for (int k = 0; k < 64; k++) s += y3[g * 64 + k] * wl[k];
    out[g] = s;
}

// ---------------- host orchestration ----------------

static inline int cdiv(long long a, int b) { return (int)((a + b - 1) / b); }

extern "C" void kernel_launch(void* const* d_in, const int* in_sizes, int n_in,
                              void* d_out, int out_size, void* d_ws, size_t ws_size,
                              hipStream_t stream) {
    const float* x     = (const float*)d_in[0];
    const int*   src   = (const int*)d_in[1];            // edge_index[0]
    const int*   dst   = ((const int*)d_in[1]) + NED;    // edge_index[1]
    const int*   batch = (const int*)d_in[2];
    // conv l: base 3 + 6l : w1,b1,w2,b2,gamma,beta
    // gru  l: base 21 + 4l : wih,whh,bih,bhh
    // proj  : base 37 : w0,b0,w1,b1,w2,b2,wl,bl
    auto F = [&](int i) { return (const float*)d_in[i]; };

    // workspace carve-up (256B aligned)
    char* ws = (char*)d_ws;
    size_t off = 0;
    auto alloc = [&](size_t bytes) { char* p = ws + off; off = (off + bytes + 255) & ~(size_t)255; return p; };
    float*          z    = (float*)alloc((size_t)NND * 128 * 4);
    float*          agg  = (float*)alloc((size_t)NND * 128 * 4);   // doubles as H
    unsigned short* Abf  = (unsigned short*)alloc((size_t)NND * 128 * 2);
    unsigned short* Rbf  = (unsigned short*)alloc((size_t)NND * 128 * 2);
    float*          stats= (float*)alloc(256 * 4);
    float*          scsh = (float*)alloc(256 * 4);
    unsigned short* cW1[3], *cW2[3], *gW[4];
    float*          gCB[4];
    for (int l = 0; l < 3; l++) { cW1[l] = (unsigned short*)alloc(16384 * 2); cW2[l] = (unsigned short*)alloc(16384 * 2); }
    for (int l = 0; l < 4; l++) { gW[l] = (unsigned short*)alloc(49152 * 2); gCB[l] = (float*)alloc(384 * 4); }
    unsigned short* pW0t = (unsigned short*)alloc(16384 * 2);
    unsigned short* pW1t = (unsigned short*)alloc(16384 * 2);
    unsigned short* pW2t = (unsigned short*)alloc(8192 * 2);
    unsigned int*   gmax = (unsigned int*)alloc((size_t)NGR * 32 * 4);
    float*          g    = (float*)alloc((size_t)NGR * 32 * 4);
    unsigned short* gbf  = (unsigned short*)alloc((size_t)NGR * 32 * 2);
    float*          gi   = (float*)alloc((size_t)NGR * 384 * 4);
    float*          hful = (float*)alloc((size_t)NGR * 128 * 4);
    unsigned short* hbf  = (unsigned short*)alloc((size_t)NGR * 128 * 2);
    unsigned short* ybf  = (unsigned short*)alloc((size_t)NGR * 128 * 2);
    unsigned short* ybf2 = (unsigned short*)alloc((size_t)NGR * 128 * 2);
    float*          y3   = (float*)alloc((size_t)NGR * 64 * 4);

    const int din_c[3]  = {128, 128, 128};
    const int dout_c[3] = {128, 128, 32};

    // ---- weight prep (bf16, [out,in] layout) ----
    for (int l = 0; l < 3; l++) {
        int b = 3 + 6 * l, di = din_c[l], dp = dout_c[l];
        k_transconv<<<cdiv((long long)di * dp, 256), 256, 0, stream>>>(F(b + 0), cW1[l], di, dp);
        k_transconv<<<cdiv((long long)dp * dp, 256), 256, 0, stream>>>(F(b + 2), cW2[l], dp, dp);
    }
    for (int l = 0; l < 4; l++) {
        int b = 21 + 4 * l, di = (l == 0) ? 32 : 128;
        k_f2bf<<<cdiv((long long)384 * di, 256), 256, 0, stream>>>(F(b + 0), gW[l], 384 * di);
        k_grubias<<<2, 256, 0, stream>>>(F(b + 2), F(b + 3), gCB[l]);
    }
    k_transconv<<<cdiv(16384, 256), 256, 0, stream>>>(F(37 + 4), pW0t, 128, 128);
    k_transconv<<<cdiv(16384, 256), 256, 0, stream>>>(F(37 + 5), pW1t, 128, 128);
    k_transconv<<<cdiv(8192, 256), 256, 0, stream>>>(F(37 + 6), pW2t, 128, 64);

    // ---- GIN conv layers ----
    const float* zin = x;
    for (int l = 0; l < 3; l++) {
        int b = 3 + 6 * l, di = din_c[l], dp = dout_c[l];
        long long nel = (long long)NND * di;
        k_copy_f32<<<cdiv(nel, 256), 256, 0, stream>>>(zin, agg, (int)nel);
        k_scatter<<<cdiv((long long)NED * 128, 256), 256, 0, stream>>>(zin, src, dst, agg);
        k_f2bf<<<cdiv(nel, 256), 256, 0, stream>>>(agg, Abf, (int)nel);
        // GEMM1: relu(agg@W1+b1) -> bf16
        k_gemm<1, 1><<<cdiv((long long)(NND / 16) * (dp / 16), 8), 256, 0, stream>>>(
            Abf, cW1[l], F(b + 1), nullptr, Rbf, NND, dp, di);
        // GEMM2: relu(R@W2+b2) -> f32 (H = agg buffer)
        k_gemm<1, 0><<<cdiv((long long)(NND / 16) * (dp / 16), 8), 256, 0, stream>>>(
            Rbf, cW2[l], F(b + 3), agg, nullptr, NND, dp, dp);
        // BatchNorm (training stats, biased var)
        k_zero_f32<<<1, 256, 0, stream>>>(stats, 256);
        k_bnreduce<<<1024, 256, 0, stream>>>(agg, stats, NND, dp);
        k_bnscale<<<1, 128, 0, stream>>>(stats, F(b + 4), F(b + 5), scsh, NND, dp);
        k_bnapply<<<cdiv((long long)NND * dp, 256), 256, 0, stream>>>(
            agg, scsh, z, (long long)NND * dp, dp - 1);
        zin = z;
    }

    // ---- global_max_pool ----
    k_fill_u32<<<cdiv(NGR * 32, 256), 256, 0, stream>>>(gmax, 0x007FFFFFu, NGR * 32);
    k_segmax<<<cdiv((long long)NND * 32, 256), 256, 0, stream>>>(z, batch, gmax, NND);
    k_gdecode<<<cdiv(NGR * 32, 256), 256, 0, stream>>>(gmax, g, gbf);

    // ---- 4-layer GRU (single step, h0 = 0) ----
    const unsigned short* hin = gbf;
    int di = 32;
    for (int l = 0; l < 4; l++) {
        int b = 21 + 4 * l;
        k_gemm<0, 0><<<cdiv((long long)(NGR / 16) * (384 / 16), 8), 256, 0, stream>>>(
            hin, gW[l], gCB[l], gi, nullptr, NGR, 384, di);
        k_gru_gate<<<cdiv(NGR * 128, 256), 256, 0, stream>>>(gi, F(b + 3), hful, hbf);
        hin = hbf; di = 128;
    }

    // ---- projection MLP ----
    k_gemm<1, 1><<<cdiv((long long)(NGR / 16) * 8, 8), 256, 0, stream>>>(
        hbf, pW0t, F(37 + 1), nullptr, ybf, NGR, 128, 128);
    k_gemm<1, 1><<<cdiv((long long)(NGR / 16) * 8, 8), 256, 0, stream>>>(
        ybf, pW1t, F(37 + 3), nullptr, ybf2, NGR, 128, 128);
    k_gemm<1, 0><<<cdiv((long long)(NGR / 16) * 4, 8), 256, 0, stream>>>(
        ybf2, pW2t, (const float*)d_in[42], y3, nullptr, NGR, 64, 128);
    k_final<<<1, 256, 0, stream>>>(y3, (const float*)d_in[43], (const float*)d_in[44],
                                   (float*)d_out);
}
